// InflatedCrossAttention_1142461301205
// MI455X (gfx1250) — matrix operand order repaired and checked by reading, not compiled, not run on hardware
//
#include <hip/hip_runtime.h>
#include <hip/hip_bf16.h>

typedef __attribute__((ext_vector_type(16))) _Float16 v16h;
typedef __attribute__((ext_vector_type(8)))  float    v8f;
typedef __attribute__((ext_vector_type(2)))  _Float16 h2;
typedef __attribute__((ext_vector_type(4)))  unsigned v4u;
typedef __attribute__((ext_vector_type(8)))  int      v8i;
typedef __attribute__((ext_vector_type(4)))  int      v4i;

#define DD 1024          // model dim (K and N of all GEMMs)
#define HEADS 16
#define DH 64
#define AB 2
#define SQ_PER_B 6144
#define SK 1024

// ---------------------------------------------------------------------------
// WMMA helper: D = A(16x32 f16) * B(32x16 f16) + C(16x16 f32)
// ---------------------------------------------------------------------------
__device__ __forceinline__ v8f wmma_f16(v16h a, v16h b, v8f c) {
  return __builtin_amdgcn_wmma_f32_16x16x32_f16(
      /*neg_a=*/false, a, /*neg_b=*/false, b,
      /*c_mod=*/(short)0, c, /*reuse_a=*/false, /*reuse_b=*/false);
}

// A-fragment (16x32, 16-bit) in this code's contiguous k order (B matches):
// lane l holds row M=l%16; VGPR v pair K = 2v + (l>=16 ? 16:0)
__device__ __forceinline__ v16h load_afrag_f32(const float* rp, int hi) {
  v16h a;
#pragma unroll
  for (int v = 0; v < 8; ++v) {
    int kk = 2 * v + (hi ? 16 : 0);
    float2 f = *(const float2*)(rp + kk);
    a[2 * v]     = (_Float16)f.x;
    a[2 * v + 1] = (_Float16)f.y;
  }
  return a;
}

__device__ __forceinline__ v16h load_afrag_f16(const _Float16* rp, int hi) {
  v16h a;
#pragma unroll
  for (int v = 0; v < 8; ++v) {
    int kk = 2 * v + (hi ? 16 : 0);
    h2 t = *(const h2*)(rp + kk);
    a[2 * v]     = t[0];
    a[2 * v + 1] = t[1];
  }
  return a;
}

// B-fragment (32x16, 16-bit): lane l holds col N=l%16;
// VGPR v pair K = 2v + (l>=16 ? 16:0). rp = row base of the N-column's
// K-contiguous data (i.e. B[k][n] = src[n*ld + k]).
__device__ __forceinline__ v16h load_bfrag_f32(const float* rp, int hi) {
  v16h b;
#pragma unroll
  for (int v = 0; v < 8; ++v) {
    float2 f = *(const float2*)(rp + 2 * v + (hi ? 16 : 0));
    b[2 * v]     = (_Float16)f.x;
    b[2 * v + 1] = (_Float16)f.y;
  }
  return b;
}

// ---------------------------------------------------------------------------
// Tensor Data Mover: async 2D tile load (f16 elements) Global -> LDS.
// Descriptor per CDNA5 ISA:
//   group0: count=1 | lds_addr[63:32] | global_addr[120:64] | type=2[127:126]
//   group1: data_size=1(2B)[17:16] | tensor_dim0[79:48] | tensor_dim1[111:80]
//           | tile_dim0[127:112] | tile_dim1[143:128] | dim0_stride[207:160]
//   groups 2/3: zero (2D tensor)
// This toolchain (clang-23 lane): 6-arg builtin
//   (uint32x4 g0, int32x8 g1, int32x4, int32x4, int32x8, i32 cpol)
// ---------------------------------------------------------------------------
__device__ __forceinline__ void tdm_load_tile_f16(
    unsigned lds_byte_addr, const void* gaddr,
    unsigned tile_w, unsigned tile_h, unsigned row_stride_elems) {
  unsigned long long ga = (unsigned long long)(uintptr_t)gaddr;
  v4u g0;
  g0[0] = 1u;                                   // count=1, user desc
  g0[1] = lds_byte_addr;                        // lds_addr
  g0[2] = (unsigned)(ga & 0xffffffffu);         // global_addr[31:0]
  g0[3] = (unsigned)((ga >> 32) & 0x1ffffffu)   // global_addr[56:32]
          | 0x80000000u;                        // type=2 at [127:126]
  v8i g1;
  g1[0] = (int)(1u << 16);                      // data_size=1 -> 2 bytes
  g1[1] = (int)((tile_w & 0xffffu) << 16);      // tensor_dim0 = tile_w
  g1[2] = (int)(((tile_w >> 16) & 0xffffu) |
                ((tile_h & 0xffffu) << 16));    // tensor_dim1 = tile_h
  g1[3] = (int)(((tile_h >> 16) & 0xffffu) |
                ((tile_w & 0xffffu) << 16));    // tile_dim0
  g1[4] = (int)(tile_h & 0xffffu);              // tile_dim1, tile_dim2=0
  g1[5] = (int)row_stride_elems;                // tensor_dim0_stride[31:0]
  g1[6] = 0;
  g1[7] = 0;
  v4i gz4 = {0, 0, 0, 0};
  v8i gz8 = {0, 0, 0, 0, 0, 0, 0, 0};
  __builtin_amdgcn_tensor_load_to_lds(g0, g1, gz4, gz4, gz8, 0);
}

// ---------------------------------------------------------------------------
// GEMM: C_f16[M,1024] = alpha * (A_f32[M,1024] @ W_f32[1024,1024]^T)
// Wave -> 16x64 tile; block = 4 waves -> 64x64 tile.
// ---------------------------------------------------------------------------
__global__ __launch_bounds__(128)
void ica_proj_gemm(const float* __restrict__ A, const float* __restrict__ W,
                   _Float16* __restrict__ C, int M, float alpha) {
  const int wave = threadIdx.x >> 5;
  const int lane = threadIdx.x & 31;
  const int hi = lane >> 4, ln = lane & 15;
  const int m0 = blockIdx.x * 64 + wave * 16;
  const int n0 = blockIdx.y * 64;

  v8f acc[4] = {};
  const float* arow = A + (size_t)(m0 + ln) * DD;

  for (int k0 = 0; k0 < DD; k0 += 32) {
    __builtin_prefetch(arow + k0 + 64, 0, 1);
    v16h a = load_afrag_f32(arow + k0, hi);
#pragma unroll
    for (int j = 0; j < 4; ++j) {
      const float* brow = W + (size_t)(n0 + j * 16 + ln) * DD + k0;
      v16h b = load_bfrag_f32(brow, hi);
      acc[j] = wmma_f16(a, b, acc[j]);
    }
  }
#pragma unroll
  for (int j = 0; j < 4; ++j)
#pragma unroll
    for (int r = 0; r < 8; ++r) {
      int row = m0 + r + 8 * hi;
      int col = n0 + j * 16 + ln;
      C[(size_t)row * DD + col] = (_Float16)(acc[j][r] * alpha);
    }
}

// ---------------------------------------------------------------------------
// Output GEMM: out_f32[M,1024] = A_f16[M,1024] @ Wo_f32[1024,1024]^T + bias
// ---------------------------------------------------------------------------
__global__ __launch_bounds__(128)
void ica_out_gemm(const _Float16* __restrict__ A, const float* __restrict__ W,
                  const float* __restrict__ bias, float* __restrict__ C, int M) {
  const int wave = threadIdx.x >> 5;
  const int lane = threadIdx.x & 31;
  const int hi = lane >> 4, ln = lane & 15;
  const int m0 = blockIdx.x * 64 + wave * 16;
  const int n0 = blockIdx.y * 64;

  v8f acc[4] = {};
  const _Float16* arow = A + (size_t)(m0 + ln) * DD;

  for (int k0 = 0; k0 < DD; k0 += 32) {
    v16h a = load_afrag_f16(arow + k0, hi);
#pragma unroll
    for (int j = 0; j < 4; ++j) {
      const float* brow = W + (size_t)(n0 + j * 16 + ln) * DD + k0;
      v16h b = load_bfrag_f32(brow, hi);
      acc[j] = wmma_f16(a, b, acc[j]);
    }
  }
#pragma unroll
  for (int j = 0; j < 4; ++j)
#pragma unroll
    for (int r = 0; r < 8; ++r) {
      int row = m0 + r + 8 * hi;
      int col = n0 + j * 16 + ln;
      C[(size_t)row * DD + col] = acc[j][r] + bias[col];
    }
}

// ---------------------------------------------------------------------------
// Flash attention: per (b,h), Sq=6144 / Sk=1024 / d=64.
// Block = 128 threads (4 waves); each wave owns 16 Q rows, block owns 64.
// K/V head data staged in LDS via the Tensor Data Mover in 128-row chunks;
// online softmax. Q was pre-scaled by dh^-0.5 in the projection kernel.
// ---------------------------------------------------------------------------
__global__ __launch_bounds__(128)
void ica_attn(const _Float16* __restrict__ q, const _Float16* __restrict__ k,
              const _Float16* __restrict__ v, _Float16* __restrict__ o) {
  __shared__ alignas(16) _Float16 Kch[128 * 64];
  __shared__ alignas(16) _Float16 Vch[128 * 64];
  __shared__ alignas(16) _Float16 Pst[4 * 16 * 32];

  const int tid = threadIdx.x;
  const int wave = tid >> 5;
  const int lane = tid & 31;
  const int hi = lane >> 4, ln = lane & 15;
  const int b = blockIdx.z, h = blockIdx.y;
  const int qb = blockIdx.x * 64 + wave * 16;

  // Q A-fragments (d 0..31 and 32..63), loaded once
  const _Float16* qrow =
      q + (size_t)(b * SQ_PER_B + qb + ln) * DD + h * DH;
  v16h aq0 = load_afrag_f16(qrow, hi);
  v16h aq1 = load_afrag_f16(qrow + 32, hi);

  v8f oacc[4] = {};
  float mrow[8], lrow[8];
#pragma unroll
  for (int r = 0; r < 8; ++r) { mrow[r] = -1e30f; lrow[r] = 0.f; }

  const _Float16* kg = k + (size_t)(b * SK) * DD + h * DH;
  const _Float16* vg = v + (size_t)(b * SK) * DD + h * DH;
  _Float16* pw = Pst + wave * 16 * 32;

  const unsigned ldsK = (unsigned)(uintptr_t)(void*)Kch;
  const unsigned ldsV = (unsigned)(uintptr_t)(void*)Vch;

  for (int c = 0; c < SK / 128; ++c) {
    __syncthreads();  // all waves done reading previous chunk
    if (wave == 0) {
      // TDM: two async 128x64 f16 tile loads (row stride = DD elems)
      tdm_load_tile_f16(ldsK, kg + (size_t)(c * 128) * DD, DH, 128, DD);
      tdm_load_tile_f16(ldsV, vg + (size_t)(c * 128) * DD, DH, 128, DD);
      __builtin_amdgcn_s_wait_tensorcnt(0);
    }
    __syncthreads();  // LDS tiles visible to all waves

#pragma unroll
    for (int kt = 0; kt < 4; ++kt) {
      const int kb = kt * 32;
      // --- scores: S[16 q, 32 k] via 2 tiles of 16 cols, K-dim = 64 ---
      v8f s_lo = {}, s_hi = {};
      {
        const _Float16* kr = Kch + (size_t)(kb + ln) * 64 + (hi ? 16 : 0);
        v16h bk;
#pragma unroll
        for (int vv = 0; vv < 8; ++vv) {
          h2 t = *(const h2*)(kr + 2 * vv);
          bk[2 * vv] = t[0]; bk[2 * vv + 1] = t[1];
        }
        s_lo = wmma_f16(aq0, bk, s_lo);
#pragma unroll
        for (int vv = 0; vv < 8; ++vv) {
          h2 t = *(const h2*)(kr + 32 + 2 * vv);
          bk[2 * vv] = t[0]; bk[2 * vv + 1] = t[1];
        }
        s_lo = wmma_f16(aq1, bk, s_lo);
      }
      {
        const _Float16* kr = Kch + (size_t)(kb + 16 + ln) * 64 + (hi ? 16 : 0);
        v16h bk;
#pragma unroll
        for (int vv = 0; vv < 8; ++vv) {
          h2 t = *(const h2*)(kr + 2 * vv);
          bk[2 * vv] = t[0]; bk[2 * vv + 1] = t[1];
        }
        s_hi = wmma_f16(aq0, bk, s_hi);
#pragma unroll
        for (int vv = 0; vv < 8; ++vv) {
          h2 t = *(const h2*)(kr + 32 + 2 * vv);
          bk[2 * vv] = t[0]; bk[2 * vv + 1] = t[1];
        }
        s_hi = wmma_f16(aq1, bk, s_hi);
      }

      // --- online softmax update (rows live across 16-lane groups) ---
#pragma unroll
      for (int r = 0; r < 8; ++r) {
        float t = fmaxf(s_lo[r], s_hi[r]);
#pragma unroll
        for (int d = 1; d < 16; d <<= 1) t = fmaxf(t, __shfl_xor(t, d, 32));
        float mn = fmaxf(mrow[r], t);
        float corr = __expf(mrow[r] - mn);
        float pl = __expf(s_lo[r] - mn);
        float ph = __expf(s_hi[r] - mn);
        float rs = pl + ph;
#pragma unroll
        for (int d = 1; d < 16; d <<= 1) rs += __shfl_xor(rs, d, 32);
        lrow[r] = lrow[r] * corr + rs;
        mrow[r] = mn;
        oacc[0][r] *= corr; oacc[1][r] *= corr;
        oacc[2][r] *= corr; oacc[3][r] *= corr;
        // stage P (C-layout -> LDS) for A-layout reload
        int prow = r + 8 * hi;
        pw[prow * 32 + ln]      = (_Float16)pl;
        pw[prow * 32 + 16 + ln] = (_Float16)ph;
      }
      // same-wave LDS ops are in-order; drain before re-reading
      asm volatile("s_wait_dscnt 0x0" ::: "memory");

      // P A-fragment (16x32)
      v16h pa;
#pragma unroll
      for (int vv = 0; vv < 8; ++vv) {
        int kk = 2 * vv + (hi ? 16 : 0);
        h2 t = *(const h2*)(pw + ln * 32 + kk);
        pa[2 * vv] = t[0]; pa[2 * vv + 1] = t[1];
      }

      // O += P @ V  (4 d-tiles of 16)
#pragma unroll
      for (int j = 0; j < 4; ++j) {
        v16h bv;
#pragma unroll
        for (int vv = 0; vv < 8; ++vv) {
          int r0 = kb + 2 * vv + (hi ? 16 : 0);
          bv[2 * vv]     = Vch[r0 * 64 + j * 16 + ln];
          bv[2 * vv + 1] = Vch[(r0 + 1) * 64 + j * 16 + ln];
        }
        oacc[j] = wmma_f16(pa, bv, oacc[j]);
      }
    }
  }

  // normalize and store f16 output at [b, qrow, h*64 + d]
#pragma unroll
  for (int r = 0; r < 8; ++r) {
    float inv = 1.0f / lrow[r];
    size_t row = (size_t)(b * SQ_PER_B + qb + r + 8 * hi);
#pragma unroll
    for (int j = 0; j < 4; ++j)
      o[row * DD + h * DH + j * 16 + ln] = (_Float16)(oacc[j][r] * inv);
  }
}

// ---------------------------------------------------------------------------
extern "C" void kernel_launch(void* const* d_in, const int* in_sizes, int n_in,
                              void* d_out, int out_size, void* d_ws, size_t ws_size,
                              hipStream_t stream) {
  (void)in_sizes; (void)n_in; (void)out_size; (void)ws_size;
  const float* x   = (const float*)d_in[0];  // [12,1024,1024] -> [12288,1024]
  const float* ctx = (const float*)d_in[1];  // [2,1024,1024]  -> [2048,1024]
  const float* Wq  = (const float*)d_in[2];
  const float* Wk  = (const float*)d_in[3];
  const float* Wv  = (const float*)d_in[4];
  const float* Wo  = (const float*)d_in[5];
  const float* bo  = (const float*)d_in[6];
  float* out = (float*)d_out;

  const int MQ = AB * SQ_PER_B;  // 12288
  const int MK = AB * SK;        // 2048

  char* ws = (char*)d_ws;
  _Float16* q_f16 = (_Float16*)ws;                 ws += (size_t)MQ * DD * 2;
  _Float16* k_f16 = (_Float16*)ws;                 ws += (size_t)MK * DD * 2;
  _Float16* v_f16 = (_Float16*)ws;                 ws += (size_t)MK * DD * 2;
  _Float16* a_f16 = (_Float16*)ws;

  const float scale = 0.125f;  // DIM_HEAD^-0.5

  // projections (scale folded into Q)
  ica_proj_gemm<<<dim3(MQ / 64, DD / 64), 128, 0, stream>>>(x,   Wq, q_f16, MQ, scale);
  ica_proj_gemm<<<dim3(MK / 64, DD / 64), 128, 0, stream>>>(ctx, Wk, k_f16, MK, 1.0f);
  ica_proj_gemm<<<dim3(MK / 64, DD / 64), 128, 0, stream>>>(ctx, Wv, v_f16, MK, 1.0f);

  // attention: grid = (Sq/64, heads, batch)
  ica_attn<<<dim3(SQ_PER_B / 64, HEADS, AB), 128, 0, stream>>>(q_f16, k_f16, v_f16, a_f16);

  // output projection + bias
  ica_out_gemm<<<dim3(MQ / 64, DD / 64), 128, 0, stream>>>(a_f16, Wo, bo, out, MQ);
}
